// MessagePassingLayer_72653666779576
// MI455X (gfx1250) — matrix operand order repaired and checked by reading, not compiled, not run on hardware
//
#include <hip/hip_runtime.h>

typedef __attribute__((ext_vector_type(16))) __bf16          v16bf;
typedef __attribute__((ext_vector_type(16))) unsigned short  v16u;
typedef __attribute__((ext_vector_type(8)))  unsigned short  v8u;
typedef __attribute__((ext_vector_type(8)))  float           v8f;

// ---------- helpers ----------

// Split an fp32 value into bf16 hi + bf16 lo (RNE), v ~= hi + lo to ~2^-16 rel.
__device__ __forceinline__ void split_bf16(float v, unsigned short& h16, unsigned short& l16) {
    unsigned u  = __float_as_uint(v);
    unsigned r  = u + 0x7FFFu + ((u >> 16) & 1u);
    unsigned short hh = (unsigned short)(r >> 16);
    float hf    = __uint_as_float((unsigned)hh << 16);
    float lo    = v - hf;
    unsigned ul = __float_as_uint(lo);
    unsigned rl = ul + 0x7FFFu + ((ul >> 16) & 1u);
    h16 = hh;
    l16 = (unsigned short)(rl >> 16);
}

// Sign-aware float atomic max via integer atomics (correct total order on IEEE bits).
__device__ __forceinline__ void atomicMaxFloat(float* addr, float value) {
    if (value >= 0.0f) {
        atomicMax((int*)addr, __float_as_int(value));
    } else {
        atomicMin((unsigned int*)addr, __float_as_uint(value));
    }
}

#define SHUF16(a, b) __builtin_shufflevector(a, b, 0,1,2,3,4,5,6,7,8,9,10,11,12,13,14,15)

// ---------- one-time packing kernels ----------

// x[N,256] fp32 -> row-major bf16 hi/lo planes
__global__ void pack_x(const float* __restrict__ x,
                       unsigned short* __restrict__ xh,
                       unsigned short* __restrict__ xl, long long n) {
    long long i = (long long)blockIdx.x * blockDim.x + threadIdx.x;
    if (i >= n) return;
    split_bf16(x[i], xh[i], xl[i]);
}

// W[256,256] fp32 (k-major rows) -> transposed bf16 hi/lo: Bt[col*256 + k]
__global__ void pack_Bt(const float* __restrict__ B,
                        unsigned short* __restrict__ bh,
                        unsigned short* __restrict__ bl) {
    int i = blockIdx.x * blockDim.x + threadIdx.x;   // 65536
    if (i >= 256 * 256) return;
    int n = i >> 8;
    int k = i & 255;
    split_bf16(B[k * 256 + n], bh[i], bl[i]);
}

// ---------- GEMM: Out[M,256] = A[M,256] @ B[256,256] via split-bf16 WMMA ----------
// A given as packed bf16 hi/lo planes (row-major), B as transposed bf16 hi/lo planes.
// mode 0: plain store (h = x @ W_gat)
// mode 1: out = relu(batchnorm(acc + b_lin)) + bias2  (feature_transform + bias_gat)
__global__ void gemm_split_bf16(const unsigned short* __restrict__ Ah,
                                const unsigned short* __restrict__ Al,
                                const unsigned short* __restrict__ Bh,
                                const unsigned short* __restrict__ Bl,
                                float* __restrict__ Out,
                                int Mrows, int n_row_tiles, int mode,
                                const float* __restrict__ b_lin,
                                const float* __restrict__ gamma,
                                const float* __restrict__ beta,
                                const float* __restrict__ mean,
                                const float* __restrict__ var,
                                const float* __restrict__ bias2) {
    const int K = 256, NC = 256;
    int wid  = (int)((blockIdx.x * blockDim.x + threadIdx.x) >> 5);
    int lane = (int)(threadIdx.x & 31);
    int row_tile = wid >> 2;      // 16-row tile
    int strip    = wid & 3;       // 64-col strip (4 x 16-col tiles)
    if (row_tile >= n_row_tiles) return;

    int m  = lane & 15;
    int hi = lane >> 4;
    int row  = row_tile * 16 + m;
    int rowc = row < Mrows ? row : (Mrows - 1);
    const unsigned short* arh = Ah + (size_t)rowc * K;
    const unsigned short* arl = Al + (size_t)rowc * K;
    int colbase = strip * 64;

    v8f acc0 = {}, acc1 = {}, acc2 = {}, acc3 = {};

#pragma unroll
    for (int kb = 0; kb < K; kb += 32) {
        // A fragment: lane holds row m; k = kb + hi*16 + j, contiguous
        int ao = kb + hi * 16;
        v8u a0 = *(const v8u*)(arh + ao);
        v8u a1 = *(const v8u*)(arh + ao + 8);
        v8u l0 = *(const v8u*)(arl + ao);
        v8u l1 = *(const v8u*)(arl + ao + 8);
        v16bf ahb = __builtin_bit_cast(v16bf, SHUF16(a0, a1));
        v16bf alb = __builtin_bit_cast(v16bf, SHUF16(l0, l1));

        // B fragment: lane holds col n; k = kb + hi*16 + j, contiguous in Bt
        int bo = kb + hi * 16;
#pragma unroll
        for (int t = 0; t < 4; ++t) {
            int col = colbase + t * 16 + m;
            v16bf bhb = __builtin_bit_cast(v16bf, *(const v16u*)(Bh + (size_t)col * K + bo));
            v16bf blb = __builtin_bit_cast(v16bf, *(const v16u*)(Bl + (size_t)col * K + bo));
            v8f c = (t == 0) ? acc0 : (t == 1) ? acc1 : (t == 2) ? acc2 : acc3;
            c = __builtin_amdgcn_wmma_f32_16x16x32_bf16(false, ahb, false, bhb, (short)0, c, false, false);
            c = __builtin_amdgcn_wmma_f32_16x16x32_bf16(false, ahb, false, blb, (short)0, c, false, false);
            c = __builtin_amdgcn_wmma_f32_16x16x32_bf16(false, alb, false, bhb, (short)0, c, false, false);
            if (t == 0) acc0 = c; else if (t == 1) acc1 = c; else if (t == 2) acc2 = c; else acc3 = c;
        }
    }

    // D layout: VGPR r, lane -> M = hi*8 + r, N = m
#pragma unroll
    for (int t = 0; t < 4; ++t) {
        int col = colbase + t * 16 + m;
        v8f c = (t == 0) ? acc0 : (t == 1) ? acc1 : (t == 2) ? acc2 : acc3;
#pragma unroll
        for (int r = 0; r < 8; ++r) {
            int orow = row_tile * 16 + hi * 8 + r;
            if (orow >= Mrows) continue;
            float v = c[r];
            if (mode == 1) {
                v += b_lin[col];
                v = (v - mean[col]) * rsqrtf(var[col] + 1e-5f) * gamma[col] + beta[col];
                v = fmaxf(v, 0.0f);
                v += bias2[col];
            }
            Out[(size_t)orow * NC + col] = v;
        }
    }
}

// ---------- per-node attention coefficients ----------
__global__ void node_attn(const float* __restrict__ h,
                          const float* __restrict__ att_src,
                          const float* __restrict__ att_dst,
                          float* __restrict__ a_src,
                          float* __restrict__ a_dst, int N) {
    int t = blockIdx.x * blockDim.x + threadIdx.x;
    if (t >= N * 4) return;
    int n = t >> 2, head = t & 3;
    const float* hp = h + (size_t)n * 256 + head * 64;
    const float* ws = att_src + head * 64;
    const float* wd = att_dst + head * 64;
    float ss = 0.f, sd = 0.f;
#pragma unroll 16
    for (int c = 0; c < 64; ++c) {
        float v = hp[c];
        ss += v * ws[c];
        sd += v * wd[c];
    }
    a_src[t] = ss;
    a_dst[t] = sd;
}

// ---------- init segment max / denom ----------
__global__ void init_maxden(float* __restrict__ amax, float* __restrict__ denom, int n) {
    int t = blockIdx.x * blockDim.x + threadIdx.x;
    if (t < n) { amax[t] = -1e30f; denom[t] = 0.0f; }
}

// ---------- edge logits + segment max ----------
__global__ void edge_logit(const int* __restrict__ ei, int E, int Etot,
                           const float* __restrict__ a_src, const float* __restrict__ a_dst,
                           float* __restrict__ logits, float* __restrict__ amax) {
    int e = blockIdx.x * blockDim.x + threadIdx.x;
    if (e >= Etot) return;
    int s, d;
    if (e < E) { s = ei[e]; d = ei[E + e]; } else { s = d = e - E; }
    float4 as = *(const float4*)(a_src + (size_t)s * 4);
    float4 ad = *(const float4*)(a_dst + (size_t)d * 4);
    float l[4] = {as.x + ad.x, as.y + ad.y, as.z + ad.z, as.w + ad.w};
#pragma unroll
    for (int hh = 0; hh < 4; ++hh) {
        float v = l[hh];
        v = v > 0.0f ? v : 0.2f * v;           // LeakyReLU(0.2)
        logits[(size_t)e * 4 + hh] = v;
        atomicMaxFloat(&amax[(size_t)d * 4 + hh], v);
    }
}

// ---------- exp(l - max) + segment sum ----------
__global__ void edge_exp(const int* __restrict__ ei, int E, int Etot,
                         const float* __restrict__ amax,
                         float* __restrict__ alpha, float* __restrict__ denom) {
    int e = blockIdx.x * blockDim.x + threadIdx.x;
    if (e >= Etot) return;
    int d = (e < E) ? ei[E + e] : (e - E);
#pragma unroll
    for (int hh = 0; hh < 4; ++hh) {
        float v = expf(alpha[(size_t)e * 4 + hh] - amax[(size_t)d * 4 + hh]);
        alpha[(size_t)e * 4 + hh] = v;
        atomicAdd(&denom[(size_t)d * 4 + hh], v);
    }
}

// ---------- weighted message scatter: out[dst] += alpha_norm * h[src] ----------
__global__ void edge_agg(const int* __restrict__ ei, int E, long long total,
                         const float* __restrict__ h,
                         const float* __restrict__ alpha,
                         const float* __restrict__ denom,
                         float* __restrict__ out) {
    long long t = (long long)blockIdx.x * blockDim.x + threadIdx.x;
    if (t >= total) return;
    int e    = (int)(t >> 6);
    int r    = (int)(t & 63);
    int head = r >> 4;
    int cg   = r & 15;
    int s, d;
    if (e < E) { s = ei[e]; d = ei[E + e]; } else { s = d = e - E; }
    float w = alpha[(size_t)e * 4 + head] / (denom[(size_t)d * 4 + head] + 1e-16f);
    float4 hv = *(const float4*)(h + (size_t)s * 256 + head * 64 + cg * 4);
    float* o = out + (size_t)d * 256 + head * 64 + cg * 4;
    atomicAdd(o + 0, hv.x * w);
    atomicAdd(o + 1, hv.y * w);
    atomicAdd(o + 2, hv.z * w);
    atomicAdd(o + 3, hv.w * w);
}

// ---------- launcher ----------
extern "C" void kernel_launch(void* const* d_in, const int* in_sizes, int n_in,
                              void* d_out, int out_size, void* d_ws, size_t ws_size,
                              hipStream_t stream) {
    const float* x        = (const float*)d_in[0];
    const int*   ei       = (const int*)d_in[1];    // [2, E] (harness passes int32)
    const float* W_gat    = (const float*)d_in[4];
    const float* att_src  = (const float*)d_in[5];
    const float* att_dst  = (const float*)d_in[6];
    const float* bias_gat = (const float*)d_in[7];
    const float* w_ft     = (const float*)d_in[12];
    const float* b_ft     = (const float*)d_in[13];
    const float* bn_gamma = (const float*)d_in[14];
    const float* bn_beta  = (const float*)d_in[15];
    const float* bn_mean  = (const float*)d_in[16];
    const float* bn_var   = (const float*)d_in[17];
    float* out = (float*)d_out;

    int N    = in_sizes[0] / 256;
    int E    = in_sizes[1] / 2;
    int Etot = E + N;

    // workspace layout
    char* ws = (char*)d_ws;
    size_t off = 0;
    float*          h     = (float*)(ws + off);          off += (size_t)N * 256 * sizeof(float);
    unsigned short* xh    = (unsigned short*)(ws + off); off += (size_t)N * 256 * sizeof(unsigned short);
    unsigned short* xl    = (unsigned short*)(ws + off); off += (size_t)N * 256 * sizeof(unsigned short);
    unsigned short* bgh   = (unsigned short*)(ws + off); off += (size_t)256 * 256 * sizeof(unsigned short);
    unsigned short* bgl   = (unsigned short*)(ws + off); off += (size_t)256 * 256 * sizeof(unsigned short);
    unsigned short* bfh   = (unsigned short*)(ws + off); off += (size_t)256 * 256 * sizeof(unsigned short);
    unsigned short* bfl   = (unsigned short*)(ws + off); off += (size_t)256 * 256 * sizeof(unsigned short);
    float* a_src  = (float*)(ws + off);                  off += (size_t)N * 4 * sizeof(float);
    float* a_dst  = (float*)(ws + off);                  off += (size_t)N * 4 * sizeof(float);
    float* amax   = (float*)(ws + off);                  off += (size_t)N * 4 * sizeof(float);
    float* denom  = (float*)(ws + off);                  off += (size_t)N * 4 * sizeof(float);
    float* alpha  = (float*)(ws + off);                  off += (size_t)Etot * 4 * sizeof(float);

    // 0) precision-split packing (x shared by both GEMMs; weights transposed)
    long long nx = (long long)N * 256;
    pack_x<<<(int)((nx + 255) / 256), 256, 0, stream>>>(x, xh, xl, nx);
    pack_Bt<<<(256 * 256 + 255) / 256, 256, 0, stream>>>(W_gat, bgh, bgl);
    pack_Bt<<<(256 * 256 + 255) / 256, 256, 0, stream>>>(w_ft, bfh, bfl);

    int nrt    = (N + 15) / 16;
    int waves  = nrt * 4;                       // 4 column strips of 64
    int blocks = (waves + 7) / 8;               // 8 waves (256 threads) per block

    // 1) h = x @ W_gat
    gemm_split_bf16<<<blocks, 256, 0, stream>>>(xh, xl, bgh, bgl, h, N, nrt, 0,
                                                nullptr, nullptr, nullptr, nullptr, nullptr, nullptr);
    // 2) out = relu(bn(x @ w_ft + b_ft)) + bias_gat   (fully initializes d_out)
    gemm_split_bf16<<<blocks, 256, 0, stream>>>(xh, xl, bfh, bfl, out, N, nrt, 1,
                                                b_ft, bn_gamma, bn_beta, bn_mean, bn_var, bias_gat);
    // 3) per-node attention coefficients
    node_attn<<<(N * 4 + 255) / 256, 256, 0, stream>>>(h, att_src, att_dst, a_src, a_dst, N);
    // 4) init segment max / denom
    init_maxden<<<(N * 4 + 255) / 256, 256, 0, stream>>>(amax, denom, N * 4);
    // 5) logits + segment max
    edge_logit<<<(Etot + 255) / 256, 256, 0, stream>>>(ei, E, Etot, a_src, a_dst, alpha, amax);
    // 6) exp + segment sum
    edge_exp<<<(Etot + 255) / 256, 256, 0, stream>>>(ei, E, Etot, amax, alpha, denom);
    // 7) weighted scatter-add into out
    long long total = (long long)Etot * 64;
    edge_agg<<<(int)((total + 255) / 256), 256, 0, stream>>>(ei, E, total, h, alpha, denom, out);
}
